// RegionProposalNetwork_4269197492384
// MI455X (gfx1250) — matrix operand order repaired and checked by reading, not compiled, not run on hardware
//
#include <hip/hip_runtime.h>
#include <hip/hip_bf16.h>
#include <stdint.h>

// ---------------- types for WMMA ----------------
typedef __bf16 v16bf __attribute__((ext_vector_type(16)));
typedef float  v8f   __attribute__((ext_vector_type(8)));

// ---------------- problem constants ----------------
#define CCH   256
#define FHW   160
#define NPIX  25600          // 160*160
#define KTOT  2304           // 9*256
#define NANCH 76800          // 3*25600
#define PRE_NMS 2000
#define TOP_K   1000
#define NEGV   (-1.0e9f)

// d_out float offsets
#define OUT_ANCH 0
#define OUT_OBJ  307200
#define OUT_DELT 384000
#define OUT_FLOG 691200
#define OUT_FBOX 692200

// ---------------- workspace layout (bytes) ----------------
static constexpr size_t WS_WHI  = 0;
static constexpr size_t WS_WLO  = WS_WHI  + (size_t)CCH * KTOT * 2;
static constexpr size_t WS_Y    = WS_WLO  + (size_t)CCH * KTOT * 2;
static constexpr size_t WS_PROP = WS_Y    + (size_t)CCH * NPIX * 4;
static constexpr size_t WS_KEYS = WS_PROP + (size_t)NANCH * 4 * 4;
static constexpr size_t WS_CTL  = WS_KEYS + (size_t)NANCH * 8;
static constexpr size_t WS_LIST = WS_CTL  + 4096;
static constexpr size_t WS_TSC  = WS_LIST + 2048 * 8;
static constexpr size_t WS_TBX  = WS_TSC  + 2048 * 4;
static constexpr size_t WS_VAL  = WS_TBX  + 2048 * 16;
static constexpr size_t WS_MASK = WS_VAL  + 2048 * 4;

struct Ctl {
    unsigned long long prefix;
    unsigned int kk;
    unsigned int cnt;
    unsigned int hist[256];
};

// ---------------- helpers ----------------
__device__ __forceinline__ unsigned short f2bf(float f) {
    unsigned int u = __float_as_uint(f);
    unsigned int r = u + 0x7FFFu + ((u >> 16) & 1u);   // RNE (finite inputs)
    return (unsigned short)(r >> 16);
}
__device__ __forceinline__ float bf2f(unsigned short b) {
    return __uint_as_float(((unsigned int)b) << 16);
}
struct Q2 { uint4 a, b; };
__device__ __forceinline__ v16bf ldfrag(const unsigned short* p0, const unsigned short* p1) {
    Q2 q;
    q.a = *reinterpret_cast<const uint4*>(p0);
    q.b = *reinterpret_cast<const uint4*>(p1);
    return __builtin_bit_cast(v16bf, q);
}

// ============================================================
// 1) weight prep: conv_w[co][ci][kh][kw] -> Whi/Wlo[co][tap*256+ci] (bf16 hi/lo)
// ============================================================
__global__ void k_prep_w(const float* __restrict__ w,
                         unsigned short* __restrict__ whi,
                         unsigned short* __restrict__ wlo) {
    int e = blockIdx.x * 256 + threadIdx.x;          // 589824 total
    int co = e / KTOT;
    int k  = e - co * KTOT;
    int tap = k >> 8;
    int ci  = k & 255;
    float v = w[co * KTOT + ci * 9 + tap];
    unsigned short h = f2bf(v);
    whi[e] = h;
    wlo[e] = f2bf(v - bf2f(h));
}

// ============================================================
// 2) conv3x3 + BN + ReLU as implicit GEMM with WMMA (bf16x2 split, f32 acc)
//    block: 256 threads = 8 waves, tile M=256 (all co) x N=64 pixels
// ============================================================
__global__ __launch_bounds__(256) void k_conv(
        const float* __restrict__ x,
        const unsigned short* __restrict__ whi,
        const unsigned short* __restrict__ wlo,
        const float* __restrict__ cb, const float* __restrict__ gam,
        const float* __restrict__ bet, const float* __restrict__ mu,
        const float* __restrict__ var, float* __restrict__ y) {
    __shared__ __align__(16) unsigned short sAh[256 * 32];
    __shared__ __align__(16) unsigned short sAl[256 * 32];
    __shared__ __align__(16) unsigned short sBh[64 * 32];
    __shared__ __align__(16) unsigned short sBl[64 * 32];

    const int tid  = threadIdx.x;
    const int lane = tid & 31;
    const int wv   = tid >> 5;
    const int wm   = (wv & 3) * 64;      // wave M base (co)
    const int wn   = (wv >> 2) * 32;     // wave N base within block
    const int n0   = blockIdx.x * 64;    // block pixel base

    v8f c[4][2];
#pragma unroll
    for (int mi = 0; mi < 4; ++mi)
#pragma unroll
        for (int ni = 0; ni < 2; ++ni)
#pragma unroll
            for (int e = 0; e < 8; ++e) c[mi][ni][e] = 0.0f;

    // B-load thread coords
    const int bn  = tid & 63;
    const int bkq = tid >> 6;            // 0..3 (8 k-values each)
    const int p   = n0 + bn;
    const int ph  = p / FHW;
    const int pw  = p - ph * FHW;

    for (int ks = 0; ks < KTOT / 32; ++ks) {
        const int k0 = ks * 32;
        __syncthreads();
        // stage A: row co = tid, 32 halves (64B) hi + lo
        {
            const uint4* g0 = reinterpret_cast<const uint4*>(whi + (size_t)tid * KTOT + k0);
            const uint4* g1 = reinterpret_cast<const uint4*>(wlo + (size_t)tid * KTOT + k0);
            uint4* d0 = reinterpret_cast<uint4*>(&sAh[tid * 32]);
            uint4* d1 = reinterpret_cast<uint4*>(&sAl[tid * 32]);
#pragma unroll
            for (int q = 0; q < 4; ++q) { d0[q] = g0[q]; d1[q] = g1[q]; }
        }
        // stage B: im2col gather, split to bf16 hi/lo (tap constant per k-step)
        {
            const int tap = k0 >> 8;
            const int dy = tap / 3 - 1, dx = tap % 3 - 1;
            const int cib = (k0 & 255) + bkq * 8;
            const int hh = ph + dy, ww = pw + dx;
            const bool ok = (hh >= 0) & (hh < FHW) & (ww >= 0) & (ww < FHW);
            unsigned short th[8], tl[8];
#pragma unroll
            for (int j = 0; j < 8; ++j) {
                float v = ok ? x[(cib + j) * NPIX + hh * FHW + ww] : 0.0f;
                unsigned short h = f2bf(v);
                th[j] = h;
                tl[j] = f2bf(v - bf2f(h));
            }
            uint4 qh, ql;
            qh.x = (unsigned)th[0] | ((unsigned)th[1] << 16);
            qh.y = (unsigned)th[2] | ((unsigned)th[3] << 16);
            qh.z = (unsigned)th[4] | ((unsigned)th[5] << 16);
            qh.w = (unsigned)th[6] | ((unsigned)th[7] << 16);
            ql.x = (unsigned)tl[0] | ((unsigned)tl[1] << 16);
            ql.y = (unsigned)tl[2] | ((unsigned)tl[3] << 16);
            ql.z = (unsigned)tl[4] | ((unsigned)tl[5] << 16);
            ql.w = (unsigned)tl[6] | ((unsigned)tl[7] << 16);
            *reinterpret_cast<uint4*>(&sBh[bn * 32 + bkq * 8]) = qh;
            *reinterpret_cast<uint4*>(&sBl[bn * 32 + bkq * 8]) = ql;
        }
        __syncthreads();

        // fragments (ISA wave32 layouts)
        const int msel = (lane >> 4) * 8;    // A: K halves 0..7 / 8..15 (+16)
        const int ksel = (lane >> 4) * 16;   // unused: B loads with msel, same K order as A
        v16bf ah[4], al[4], bh[2], bl[2];
#pragma unroll
        for (int mi = 0; mi < 4; ++mi) {
            int row = (wm + mi * 16 + (lane & 15)) * 32;
            ah[mi] = ldfrag(&sAh[row + msel], &sAh[row + 16 + msel]);
            al[mi] = ldfrag(&sAl[row + msel], &sAl[row + 16 + msel]);
        }
#pragma unroll
        for (int ni = 0; ni < 2; ++ni) {
            int row = (wn + ni * 16 + (lane & 15)) * 32;
            bh[ni] = ldfrag(&sBh[row + msel], &sBh[row + 16 + msel]);
            bl[ni] = ldfrag(&sBl[row + msel], &sBl[row + 16 + msel]);
        }
#pragma unroll
        for (int mi = 0; mi < 4; ++mi)
#pragma unroll
            for (int ni = 0; ni < 2; ++ni) {
                c[mi][ni] = __builtin_amdgcn_wmma_f32_16x16x32_bf16(
                    false, ah[mi], false, bh[ni], (short)0, c[mi][ni], false, false);
                c[mi][ni] = __builtin_amdgcn_wmma_f32_16x16x32_bf16(
                    false, ah[mi], false, bl[ni], (short)0, c[mi][ni], false, false);
                c[mi][ni] = __builtin_amdgcn_wmma_f32_16x16x32_bf16(
                    false, al[mi], false, bh[ni], (short)0, c[mi][ni], false, false);
            }
    }

    // epilogue: bias + BN(eval) + ReLU -> y[co][p]
#pragma unroll
    for (int mi = 0; mi < 4; ++mi)
#pragma unroll
        for (int ni = 0; ni < 2; ++ni) {
            int pp = n0 + wn + ni * 16 + (lane & 15);
#pragma unroll
            for (int e = 0; e < 8; ++e) {
                int co = wm + mi * 16 + (lane >> 4) * 8 + e;
                float s = gam[co] * rsqrtf(var[co] + 1e-5f);
                float v = (c[mi][ni][e] + cb[co] - mu[co]) * s + bet[co];
                y[co * NPIX + pp] = v > 0.0f ? v : 0.0f;
            }
        }
}

// ============================================================
// 3) 1x1 heads: obj (3) + reg (12), write into d_out in RPN layout
// ============================================================
__global__ void k_heads(const float* __restrict__ y,
                        const float* __restrict__ dw, const float* __restrict__ db,
                        const float* __restrict__ rw, const float* __restrict__ rb,
                        float* __restrict__ out) {
    int p = blockIdx.x * 256 + threadIdx.x;   // 25600
    float ad[3] = {0.f, 0.f, 0.f};
    float ar[12];
#pragma unroll
    for (int o = 0; o < 12; ++o) ar[o] = 0.f;
    for (int co = 0; co < CCH; ++co) {
        float yv = y[co * NPIX + p];
#pragma unroll
        for (int o = 0; o < 3; ++o)  ad[o] += dw[o * CCH + co] * yv;
#pragma unroll
        for (int o = 0; o < 12; ++o) ar[o] += rw[o * CCH + co] * yv;
    }
#pragma unroll
    for (int a = 0; a < 3; ++a)
        out[OUT_OBJ + a * NPIX + p] = ad[a] + db[a];
#pragma unroll
    for (int c4 = 0; c4 < 4; ++c4)
#pragma unroll
        for (int a = 0; a < 3; ++a)
            out[OUT_DELT + (size_t)(a * NPIX + p) * 4 + c4] = ar[c4 * 3 + a] + rb[c4 * 3 + a];
}

// ============================================================
// 4) anchors + proposal decode + clipping + masked score key
// ============================================================
__global__ void k_prop(float* __restrict__ out, float* __restrict__ props,
                       unsigned long long* __restrict__ keys) {
    int i = blockIdx.x * 256 + threadIdx.x;   // 76800
    int a = i / NPIX;
    int p = i - a * NPIX;
    int h = p / FHW;
    int w = p - h * FHW;
    const float sq[3] = {0.70710678118654752f, 1.0f, 1.41421356237309505f};
    float ha = 64.0f * sq[a], wa = 64.0f / sq[a];
    float sx = (float)w * 4.0f, sy = (float)h * 4.0f;
    float x1 = fminf(fmaxf(sx - 0.5f * wa, 0.0f), 640.0f);
    float y1 = fminf(fmaxf(sy - 0.5f * ha, 0.0f), 640.0f);
    float x2 = fminf(fmaxf(sx + 0.5f * wa, 0.0f), 640.0f);
    float y2 = fminf(fmaxf(sy + 0.5f * ha, 0.0f), 640.0f);
    out[OUT_ANCH + (size_t)i * 4 + 0] = x1;
    out[OUT_ANCH + (size_t)i * 4 + 1] = y1;
    out[OUT_ANCH + (size_t)i * 4 + 2] = x2;
    out[OUT_ANCH + (size_t)i * 4 + 3] = y2;

    float aw = x2 - x1, ah2 = y2 - y1;
    float cx = x1 + 0.5f * aw, cy = y1 + 0.5f * ah2;
    float d0 = out[OUT_DELT + (size_t)i * 4 + 0];
    float d1 = out[OUT_DELT + (size_t)i * 4 + 1];
    float d2 = out[OUT_DELT + (size_t)i * 4 + 2];
    float d3 = out[OUT_DELT + (size_t)i * 4 + 3];
    float pcx = d0 * aw + cx, pcy = d1 * ah2 + cy;
    float pw2 = aw * expf(d2), ph2 = ah2 * expf(d3);
    float px1 = fminf(fmaxf(pcx - 0.5f * pw2, 0.0f), 640.0f);
    float py1 = fminf(fmaxf(pcy - 0.5f * ph2, 0.0f), 640.0f);
    float px2 = fminf(fmaxf(pcx + 0.5f * pw2, 0.0f), 640.0f);
    float py2 = fminf(fmaxf(pcy + 0.5f * ph2, 0.0f), 640.0f);
    props[(size_t)i * 4 + 0] = px1;
    props[(size_t)i * 4 + 1] = py1;
    props[(size_t)i * 4 + 2] = px2;
    props[(size_t)i * 4 + 3] = py2;

    float lg = out[OUT_OBJ + i];
    float masked = lg > 0.0f ? lg : NEGV;     // sigmoid(lg)>0.5 <=> lg>0
    unsigned int fb = __float_as_uint(masked);
    unsigned int s  = (fb & 0x80000000u) ? ~fb : (fb | 0x80000000u);
    keys[i] = ((unsigned long long)s << 32) | (unsigned long long)(0xFFFFFFFFu - (unsigned)i);
}

// ============================================================
// 5) init: fill filt outputs defaults + Ctl
// ============================================================
__global__ void k_init(float* __restrict__ out, Ctl* __restrict__ ctl) {
    int g = blockIdx.x * 256 + threadIdx.x;   // 24*256 = 6144
    if (g < TOP_K) out[OUT_FLOG + g] = NEGV;
    else if (g < 5000) out[OUT_FBOX + (g - 1000)] = 0.0f;
    if (g == 5000) { ctl->prefix = 0ull; ctl->kk = PRE_NMS; ctl->cnt = 0u; }
    if (g >= 5008 && g < 5264) ctl->hist[g - 5008] = 0u;
}

// ============================================================
// 6) radix-select (8-bit digits, MSB-first) over unique 64-bit keys
// ============================================================
__global__ void k_hist(const unsigned long long* __restrict__ keys,
                       Ctl* __restrict__ ctl, int r) {
    int i = blockIdx.x * 256 + threadIdx.x;
    unsigned long long k = keys[i];
    bool ok = (r == 0) || ((k >> (64 - 8 * r)) == ctl->prefix);
    if (ok) atomicAdd(&ctl->hist[(unsigned)(k >> (56 - 8 * r)) & 255u], 1u);
}

__global__ void k_pick(Ctl* __restrict__ ctl) {
    unsigned int kk = ctl->kk;
    unsigned int cum = 0;
    int d = 255;
    for (; d >= 0; --d) {
        unsigned int c = ctl->hist[d];
        if (cum + c >= kk) break;
        cum += c;
    }
    ctl->prefix = (ctl->prefix << 8) | (unsigned long long)(unsigned)d;
    ctl->kk = kk - cum;
    for (int j = 0; j < 256; ++j) ctl->hist[j] = 0u;
}

__global__ void k_compact(const unsigned long long* __restrict__ keys,
                          Ctl* __restrict__ ctl,
                          unsigned long long* __restrict__ list) {
    int i = blockIdx.x * 256 + threadIdx.x;
    unsigned long long k = keys[i];
    if (k >= ctl->prefix) {                      // exactly PRE_NMS keys (unique)
        unsigned int pos = atomicAdd(&ctl->cnt, 1u);
        if (pos < PRE_NMS) list[pos] = k;
    }
}

// ============================================================
// 7) single-block bitonic sort (2048, descending) + gather top boxes/scores
// ============================================================
__global__ __launch_bounds__(1024) void k_sort(
        const unsigned long long* __restrict__ list,
        const float* __restrict__ props,
        float* __restrict__ tsc, float* __restrict__ tbx,
        unsigned int* __restrict__ valid) {
    __shared__ unsigned long long s[2048];
    int t = threadIdx.x;
    s[t]        = (t < PRE_NMS) ? list[t] : 0ull;
    s[t + 1024] = (t + 1024 < PRE_NMS) ? list[t + 1024] : 0ull;
    __syncthreads();
    for (int size = 2; size <= 2048; size <<= 1) {
        for (int stride = size >> 1; stride > 0; stride >>= 1) {
            for (int i = t; i < 2048; i += 1024) {
                int j = i ^ stride;
                if (j > i) {
                    unsigned long long a = s[i], b = s[j];
                    bool desc = ((i & size) == 0);
                    if ((a < b) == desc) { s[i] = b; s[j] = a; }
                }
            }
            __syncthreads();
        }
    }
    for (int idx = t; idx < PRE_NMS; idx += 1024) {
        unsigned long long k = s[idx];
        unsigned int id = 0xFFFFFFFFu - (unsigned int)(k & 0xFFFFFFFFull);
        unsigned int sb = (unsigned int)(k >> 32);
        unsigned int fb = (sb & 0x80000000u) ? (sb ^ 0x80000000u) : ~sb;
        float sc = __uint_as_float(fb);
        tsc[idx] = sc;
        valid[idx] = (sc > NEGV * 0.5f) ? 1u : 0u;
#pragma unroll
        for (int cc = 0; cc < 4; ++cc)
            tbx[(size_t)idx * 4 + cc] = props[(size_t)id * 4 + cc];
    }
}

// ============================================================
// 8) suppression bitmask: mask[i][w32] over j (iou>0.7 && j>i)
// ============================================================
__global__ void k_mask(const float* __restrict__ tbx, unsigned int* __restrict__ mask) {
    int i = blockIdx.x;
    float xi1 = tbx[i * 4 + 0], yi1 = tbx[i * 4 + 1];
    float xi2 = tbx[i * 4 + 2], yi2 = tbx[i * 4 + 3];
    float ai = (xi2 - xi1) * (yi2 - yi1);
    for (int j = threadIdx.x; j < 2048; j += 256) {
        bool sup = false;
        if (j < PRE_NMS && j > i) {
            float xj1 = tbx[j * 4 + 0], yj1 = tbx[j * 4 + 1];
            float xj2 = tbx[j * 4 + 2], yj2 = tbx[j * 4 + 3];
            float aj = (xj2 - xj1) * (yj2 - yj1);
            float lx = fmaxf(xi1, xj1), ly = fmaxf(yi1, yj1);
            float rx = fminf(xi2, xj2), ry = fminf(yi2, yj2);
            float w = fmaxf(rx - lx, 0.0f), h = fmaxf(ry - ly, 0.0f);
            float inter = w * h;
            float iou = inter / (ai + aj - inter + 1e-9f);
            sup = iou > 0.7f;
        }
        unsigned long long bal = __ballot(sup);
        if ((threadIdx.x & 31) == 0) mask[i * 64 + (j >> 5)] = (unsigned int)bal;
    }
}

// ============================================================
// 9) sequential greedy NMS scan (1 wave) + write final 1000 slots
// ============================================================
__global__ void k_scan(const unsigned int* __restrict__ mask,
                       const unsigned int* __restrict__ valid,
                       const float* __restrict__ tsc, const float* __restrict__ tbx,
                       float* __restrict__ out) {
    __shared__ unsigned int rem[64];
    int l = threadIdx.x;   // 32 threads
    for (int w = l; w < 64; w += 32) {
        unsigned int bits = 0;
        for (int b = 0; b < 32; ++b) {
            int t = w * 32 + b;
            unsigned int v = (t < PRE_NMS) ? valid[t] : 0u;
            if (!v) bits |= (1u << b);
        }
        rem[w] = bits;
    }
    __syncthreads();
    unsigned int count = 0;
    for (int i = 0; i < PRE_NMS; ++i) {
        unsigned int word = rem[i >> 5];
        bool alive = ((word >> (i & 31)) & 1u) == 0u;
        __syncthreads();
        if (alive) {
            if (count < TOP_K) {
                if (l == 0) out[OUT_FLOG + count] = tsc[i];
                if (l < 4)  out[OUT_FBOX + count * 4 + l] = tbx[i * 4 + l];
            }
            rem[l]      |= mask[i * 64 + l];
            rem[l + 32] |= mask[i * 64 + 32 + l];
            count++;
        }
        __syncthreads();
    }
}

// ============================================================
// launcher
// ============================================================
extern "C" void kernel_launch(void* const* d_in, const int* in_sizes, int n_in,
                              void* d_out, int out_size, void* d_ws, size_t ws_size,
                              hipStream_t stream) {
    const float* x      = (const float*)d_in[0];
    const float* conv_w = (const float*)d_in[1];
    const float* conv_b = (const float*)d_in[2];
    const float* gam    = (const float*)d_in[3];
    const float* bet    = (const float*)d_in[4];
    const float* mu     = (const float*)d_in[5];
    const float* var    = (const float*)d_in[6];
    const float* det_w  = (const float*)d_in[7];
    const float* det_b  = (const float*)d_in[8];
    const float* reg_w  = (const float*)d_in[9];
    const float* reg_b  = (const float*)d_in[10];
    float* out = (float*)d_out;

    char* ws = (char*)d_ws;
    unsigned short* whi = (unsigned short*)(ws + WS_WHI);
    unsigned short* wlo = (unsigned short*)(ws + WS_WLO);
    float* y            = (float*)(ws + WS_Y);
    float* props        = (float*)(ws + WS_PROP);
    unsigned long long* keys = (unsigned long long*)(ws + WS_KEYS);
    Ctl* ctl            = (Ctl*)(ws + WS_CTL);
    unsigned long long* list = (unsigned long long*)(ws + WS_LIST);
    float* tsc          = (float*)(ws + WS_TSC);
    float* tbx          = (float*)(ws + WS_TBX);
    unsigned int* valid = (unsigned int*)(ws + WS_VAL);
    unsigned int* mask  = (unsigned int*)(ws + WS_MASK);

    k_prep_w<<<KTOT, 256, 0, stream>>>(conv_w, whi, wlo);
    k_conv<<<NPIX / 64, 256, 0, stream>>>(x, whi, wlo, conv_b, gam, bet, mu, var, y);
    k_heads<<<NPIX / 256, 256, 0, stream>>>(y, det_w, det_b, reg_w, reg_b, out);
    k_prop<<<NANCH / 256, 256, 0, stream>>>(out, props, keys);
    k_init<<<24, 256, 0, stream>>>(out, ctl);
    for (int r = 0; r < 8; ++r) {
        k_hist<<<NANCH / 256, 256, 0, stream>>>(keys, ctl, r);
        k_pick<<<1, 1, 0, stream>>>(ctl);
    }
    k_compact<<<NANCH / 256, 256, 0, stream>>>(keys, ctl, list);
    k_sort<<<1, 1024, 0, stream>>>(list, props, tsc, tbx, valid);
    k_mask<<<PRE_NMS, 256, 0, stream>>>(tbx, mask);
    k_scan<<<1, 32, 0, stream>>>(mask, valid, tsc, tbx, out);
}
